// LSTMWithGates_44521630990411
// MI455X (gfx1250) — matrix operand order repaired and checked by reading, not compiled, not run on hardware
//
#include <hip/hip_runtime.h>
#include <hip/hip_bf16.h>

typedef __bf16 bf16_t;
typedef __bf16 v16bf __attribute__((ext_vector_type(16)));
typedef __bf16 v4bf  __attribute__((ext_vector_type(4)));
typedef float  v8f   __attribute__((ext_vector_type(8)));
typedef float  v4f   __attribute__((ext_vector_type(4)));
typedef int    v4i   __attribute__((ext_vector_type(4)));

#define HID    512
#define NBATCH 64
#define SEQ    512
#define G4     2048   // 4*HID
#define NWG    32     // workgroups in recurrent kernel

// ---------------------------------------------------------------------------
// Fragment helper: 8 VGPRs (16 bf16) assembled from two 16-byte chunks.
// Callers use contiguous K order, k = 16*half + s, for A (v0..3 = first chunk, v4..7 = second) and
// B alike (one contiguous 32B run of K per lane-half); not the ISA table order, but consistent.
// ---------------------------------------------------------------------------
union FragU {
    v16bf bf;
    struct { v4i lo; v4i hi; } p;
};

__device__ __forceinline__ v16bf make_frag(const bf16_t* p0, const bf16_t* p1) {
    FragU u;
    u.p.lo = *reinterpret_cast<const v4i*>(p0);
    u.p.hi = *reinterpret_cast<const v4i*>(p1);
    return u.bf;
}

__device__ __forceinline__ float sigmoid_fast(float x) {
    return 1.0f / (1.0f + __expf(-x));
}
__device__ __forceinline__ float tanh_fast(float x) {
    float e = __expf(-2.0f * x);
    return (1.0f - e) / (1.0f + e);
}

// ---------------------------------------------------------------------------
// fp32 -> bf16 conversion (vectorized by 4)
// ---------------------------------------------------------------------------
__global__ void cvt_bf16_kernel(const float* __restrict__ src,
                                bf16_t* __restrict__ dst, int n4) {
    int i = blockIdx.x * blockDim.x + threadIdx.x;
    if (i < n4) {
        v4f v = reinterpret_cast<const v4f*>(src)[i];
        v4bf o;
        o.x = (bf16_t)v.x; o.y = (bf16_t)v.y;
        o.z = (bf16_t)v.z; o.w = (bf16_t)v.w;
        reinterpret_cast<v4bf*>(dst)[i] = o;
    }
}

// ---------------------------------------------------------------------------
// Phase 1: x_proj[r, c] = sum_k x[r,k] * W_ih[c,k] + b_ih[c] + b_hh[c]
// rows R = B*S = 32768, cols = 4H = 2048, K = 512.
// Block: 256 threads = 8 waves (2x4), 64(M) x 256(N) tile, K-step 32.
// Wave tile 32x64 -> 8 WMMAs per k-step. Software-pipelined global->LDS.
// ---------------------------------------------------------------------------
__global__ __launch_bounds__(256) void xproj_gemm_kernel(
    const bf16_t* __restrict__ xbf, const bf16_t* __restrict__ wbf,
    const float* __restrict__ b_ih, const float* __restrict__ b_hh,
    float* __restrict__ xproj) {
    __shared__ bf16_t Xl[64][32];    //  4 KB
    __shared__ bf16_t Wl[256][32];   // 16 KB

    const int t    = threadIdx.x;
    const int r0   = blockIdx.y * 64;
    const int c0   = blockIdx.x * 256;
    const int w    = t >> 5, lane = t & 31;
    const int wm   = w >> 2, wn   = w & 3;
    const int lr   = lane & 15, half = lane >> 4;

    const int xrow = t >> 2, xch = t & 3;   // X tile: 16 B per thread
    // W tile: 64 B per thread, row = t

    v4i xreg, wreg0, wreg1, wreg2, wreg3;
    auto gload = [&](int kb) {
        xreg = *reinterpret_cast<const v4i*>(
            xbf + (size_t)(r0 + xrow) * HID + kb + xch * 8);
        const v4i* s = reinterpret_cast<const v4i*>(
            wbf + (size_t)(c0 + t) * HID + kb);
        wreg0 = s[0]; wreg1 = s[1]; wreg2 = s[2]; wreg3 = s[3];
    };

    v8f acc[2][4] = {};
    gload(0);
    for (int kb = 0; kb < HID; kb += 32) {
        __syncthreads();
        *reinterpret_cast<v4i*>(&Xl[xrow][xch * 8]) = xreg;
        {
            v4i* d = reinterpret_cast<v4i*>(&Wl[t][0]);
            d[0] = wreg0; d[1] = wreg1; d[2] = wreg2; d[3] = wreg3;
        }
        __syncthreads();
        if (kb + 32 < HID) gload(kb + 32);   // overlap next loads with WMMA

        v16bf afrag[2], bfrag[4];
        #pragma unroll
        for (int mt = 0; mt < 2; ++mt) {
            const bf16_t* ap = &Xl[wm * 32 + mt * 16 + lr][0];
            afrag[mt] = make_frag(ap + half * 16, ap + half * 16 + 8);
        }
        #pragma unroll
        for (int nt = 0; nt < 4; ++nt) {
            const bf16_t* bp = &Wl[wn * 64 + nt * 16 + lr][half * 16];
            bfrag[nt] = make_frag(bp, bp + 8);
        }
        #pragma unroll
        for (int mt = 0; mt < 2; ++mt)
            #pragma unroll
            for (int nt = 0; nt < 4; ++nt)
                acc[mt][nt] = __builtin_amdgcn_wmma_f32_16x16x32_bf16(
                    false, afrag[mt], false, bfrag[nt],
                    (short)0, acc[mt][nt], false, false);
    }

    // epilogue: add biases, streamed (non-temporal) store
    #pragma unroll
    for (int mt = 0; mt < 2; ++mt)
        #pragma unroll
        for (int nt = 0; nt < 4; ++nt) {
            int   cg    = c0 + wn * 64 + nt * 16 + lr;
            float bias  = b_ih[cg] + b_hh[cg];
            int   rbase = r0 + wm * 32 + mt * 16 + half * 8;
            #pragma unroll
            for (int v = 0; v < 8; ++v)
                __builtin_nontemporal_store(acc[mt][nt][v] + bias,
                    &xproj[(size_t)(rbase + v) * G4 + cg]);
        }
}

// ---------------------------------------------------------------------------
// Phase 2: persistent recurrent kernel. 32 WGs; WG g owns hidden columns
// [g*16, g*16+16) and gate columns {gate*512 + g*16 ..} for gate=0..3.
// W_hh slice (64 KB) staged once into LDS; per step: 16 WMMA tiles
// (4 m-tiles x 4 gates) over K=512, gates -> LDS, vectorized elementwise
// c/h update with c in registers, device-scope barrier between steps.
// ---------------------------------------------------------------------------
__global__ __launch_bounds__(256) void lstm_recurrent_kernel(
    const float* __restrict__ xproj, const bf16_t* __restrict__ whhbf,
    bf16_t* __restrict__ hbuf, float* __restrict__ out,
    unsigned* __restrict__ counter) {
    __shared__ bf16_t Wlds[64][HID]; // 64 KB: row = gate*16 + slice col
    __shared__ float  gl[4][64][16]; // 16 KB gate staging

    const int g    = blockIdx.x;               // column-slice owner
    const int t    = threadIdx.x;
    const int w    = t >> 5, lane = t & 31;
    const int lr   = lane & 15, half = lane >> 4;
    const int mt   = w & 3;                    // m-tile 0..3
    const int gp   = w >> 2;                   // gate pair 0..1
    const int g0   = 2 * gp, g1 = 2 * gp + 1;

    // stage this WG's W_hh slice into LDS (4096 x 16B chunks, coalesced)
    #pragma unroll
    for (int i = 0; i < 16; ++i) {
        int idx = i * 256 + t;
        int r   = idx >> 6;        // LDS row 0..63
        int ch  = idx & 63;        // 16B chunk within row
        int grow = (r >> 4) * HID + g * 16 + (r & 15);  // W_hh row
        *reinterpret_cast<v4i*>(&Wlds[r][ch * 8]) =
            *reinterpret_cast<const v4i*>(whhbf + (size_t)grow * HID + ch * 8);
    }

    float c[4] = {0.f, 0.f, 0.f, 0.f};
    __syncthreads();

    for (int step = 0; step < SEQ; ++step) {
        const bf16_t* hread = hbuf + (size_t)(step & 1) * NBATCH * HID;

        v8f acc0 = {}, acc1 = {};
        #pragma unroll 4
        for (int kb = 0; kb < HID; kb += 32) {
            const bf16_t* ap = hread + (size_t)(mt * 16 + lr) * HID + kb;
            v16bf a = make_frag(ap + half * 16, ap + half * 16 + 8);
            const bf16_t* bp0 = &Wlds[g0 * 16 + lr][kb + half * 16];
            const bf16_t* bp1 = &Wlds[g1 * 16 + lr][kb + half * 16];
            v16bf b0 = make_frag(bp0, bp0 + 8);
            v16bf b1 = make_frag(bp1, bp1 + 8);
            acc0 = __builtin_amdgcn_wmma_f32_16x16x32_bf16(
                false, a, false, b0, (short)0, acc0, false, false);
            acc1 = __builtin_amdgcn_wmma_f32_16x16x32_bf16(
                false, a, false, b1, (short)0, acc1, false, false);
        }

        #pragma unroll
        for (int v = 0; v < 8; ++v) {
            gl[g0][mt * 16 + half * 8 + v][lr] = acc0[v];
            gl[g1][mt * 16 + half * 8 + v][lr] = acc1[v];
        }
        __syncthreads();

        // elementwise: thread owns batch row b = t>>2, 4 contiguous cols
        const int    b      = t >> 2;
        const int    hc0    = (t & 3) * 4;
        const size_t xpbase = ((size_t)b * SEQ + step) * G4 + g * 16 + hc0;
        const size_t obase  = ((size_t)b * SEQ + step) * HID + g * 16 + hc0;
        bf16_t* hw = hbuf + (size_t)((step + 1) & 1) * NBATCH * HID
                   + (size_t)b * HID + g * 16 + hc0;

        v4f ig4 = *reinterpret_cast<const v4f*>(&gl[0][b][hc0]);
        v4f fg4 = *reinterpret_cast<const v4f*>(&gl[1][b][hc0]);
        v4f gg4 = *reinterpret_cast<const v4f*>(&gl[2][b][hc0]);
        v4f og4 = *reinterpret_cast<const v4f*>(&gl[3][b][hc0]);
        v4f xi = __builtin_nontemporal_load(
            reinterpret_cast<const v4f*>(&xproj[xpbase + 0 * HID]));
        v4f xf = __builtin_nontemporal_load(
            reinterpret_cast<const v4f*>(&xproj[xpbase + 1 * HID]));
        v4f xg = __builtin_nontemporal_load(
            reinterpret_cast<const v4f*>(&xproj[xpbase + 2 * HID]));
        v4f xo = __builtin_nontemporal_load(
            reinterpret_cast<const v4f*>(&xproj[xpbase + 3 * HID]));

        v4f  hv;
        v4bf hb;
        #pragma unroll
        for (int j = 0; j < 4; ++j) {
            float ia = sigmoid_fast(ig4[j] + xi[j]);
            float fa = sigmoid_fast(fg4[j] + xf[j]);
            float ga = tanh_fast(gg4[j] + xg[j]);
            float oa = sigmoid_fast(og4[j] + xo[j]);
            c[j] = fa * c[j] + ia * ga;
            float h = oa * tanh_fast(c[j]);
            hv[j] = h;
            hb[j] = (bf16_t)h;
        }
        __builtin_nontemporal_store(hv, reinterpret_cast<v4f*>(&out[obase]));
        *reinterpret_cast<v4bf*>(hw) = hb;

        if (step + 1 < SEQ)
            __builtin_prefetch(&xproj[xpbase + G4], 0, 0);  // next timestep row

        // device-wide barrier: arrive (release) + spin (acquire) on one lane
        __threadfence();
        __syncthreads();
        if (t == 0) {
            __hip_atomic_fetch_add(counter, 1u, __ATOMIC_RELEASE,
                                   __HIP_MEMORY_SCOPE_AGENT);
            const unsigned target = (unsigned)NWG * (unsigned)(step + 1);
            while (__hip_atomic_load(counter, __ATOMIC_ACQUIRE,
                                     __HIP_MEMORY_SCOPE_AGENT) < target)
                __builtin_amdgcn_s_sleep(1);
        }
        __syncthreads();
    }
}

// ---------------------------------------------------------------------------
// Host entry
// ---------------------------------------------------------------------------
extern "C" void kernel_launch(void* const* d_in, const int* in_sizes, int n_in,
                              void* d_out, int out_size, void* d_ws, size_t ws_size,
                              hipStream_t stream) {
    const float* x   = (const float*)d_in[0];   // [64, 512, 512]
    const float* Wih = (const float*)d_in[1];   // [2048, 512]
    const float* Whh = (const float*)d_in[2];   // [2048, 512]
    const float* bih = (const float*)d_in[3];   // [2048]
    const float* bhh = (const float*)d_in[4];   // [2048]
    float* out = (float*)d_out;                 // [64, 512, 512]

    // workspace layout (256-aligned)
    char* ws = (char*)d_ws;
    const size_t CNT_OFF  = 0;                                   // 256 B
    const size_t HBUF_OFF = 256;                                 // 2*64*512*2 = 128 KB
    const size_t XBF_OFF  = HBUF_OFF + 2 * NBATCH * HID * 2;     // 32 MB
    const size_t WIH_OFF  = XBF_OFF + (size_t)NBATCH * SEQ * HID * 2;
    const size_t WHH_OFF  = WIH_OFF + (size_t)G4 * HID * 2;
    const size_t XPRJ_OFF = WHH_OFF + (size_t)G4 * HID * 2;      // 256 MB fp32

    unsigned* counter = (unsigned*)(ws + CNT_OFF);
    bf16_t*   hbuf    = (bf16_t*)(ws + HBUF_OFF);
    bf16_t*   xbf     = (bf16_t*)(ws + XBF_OFF);
    bf16_t*   wihbf   = (bf16_t*)(ws + WIH_OFF);
    bf16_t*   whhbf   = (bf16_t*)(ws + WHH_OFF);
    float*    xproj   = (float*)(ws + XPRJ_OFF);

    // zero barrier counter + h0/h1 buffers (graph-capture safe)
    hipMemsetAsync(ws, 0, HBUF_OFF + 2 * NBATCH * HID * 2, stream);

    // fp32 -> bf16 conversions
    {
        int n4 = NBATCH * SEQ * HID / 4;   // 4,194,304
        cvt_bf16_kernel<<<n4 / 256, 256, 0, stream>>>(x, xbf, n4);
    }
    {
        int n4 = G4 * HID / 4;             // 262,144
        cvt_bf16_kernel<<<n4 / 256, 256, 0, stream>>>(Wih, wihbf, n4);
        cvt_bf16_kernel<<<n4 / 256, 256, 0, stream>>>(Whh, whhbf, n4);
    }

    // Phase 1: x_proj GEMM (32768 x 2048 x 512)
    xproj_gemm_kernel<<<dim3(G4 / 256, (NBATCH * SEQ) / 64), 256, 0, stream>>>(
        xbf, wihbf, bih, bhh, xproj);

    // Phase 2: persistent recurrence (32 co-resident WGs)
    lstm_recurrent_kernel<<<NWG, 256, 0, stream>>>(xproj, whhbf, hbuf, out, counter);
}
